// PatchSelfAttentionHead_4698694222309
// MI455X (gfx1250) — matrix operand order repaired and checked by reading, not compiled, not run on hardware
//
#include <hip/hip_runtime.h>
#include <hip/hip_bf16.h>

// ---------------- problem constants ----------------
static constexpr int kB   = 16;
static constexpr int kT   = 512;   // task tokens
static constexpr int kD   = 256;
static constexpr int kNI  = 85;
static constexpr int kNT  = 77;
static constexpr int kL   = 2;
static constexpr int kFF  = 1024;
static constexpr int kTH  = 32;
static constexpr int kNH  = 8;
static constexpr int kHD  = 32;             // head dim
static constexpr int kS   = kT + kNI + kNT; // 674
static constexpr int kSP  = 704;            // padded seq (multiple of 32)
static constexpr int kM   = kB * kSP;       // 11264 rows in the big token matrix

typedef _Float16 v16h __attribute__((ext_vector_type(16)));
typedef _Float16 v8h  __attribute__((ext_vector_type(8)));
typedef float    v8f  __attribute__((ext_vector_type(8)));

// ---------------- helpers ----------------
__device__ __forceinline__ v16h load_a_frag(const _Float16* row, int half) {
  // A 16x32 f16 fragment: lane m=l&15; half0 -> K {0..15}, half1 -> K {16..31}
  v16h a;
  const int base = half ? 16 : 0;
  ((v8h*)&a)[0] = *(const v8h*)(row + base);
  ((v8h*)&a)[1] = *(const v8h*)(row + base + 8);
  return a;
}

__device__ __forceinline__ float block_reduce_sum_256(float v, float* sbuf) {
  #pragma unroll
  for (int m = 16; m >= 1; m >>= 1) v += __shfl_xor(v, m, 32);
  const int w = threadIdx.x >> 5;
  if ((threadIdx.x & 31) == 0) sbuf[w] = v;
  __syncthreads();
  float r = sbuf[0];
  #pragma unroll
  for (int i = 1; i < 8; ++i) r += sbuf[i];
  __syncthreads();
  return r;
}

// ---------------- small prep kernels ----------------
__global__ void cvt_f32_f16_kernel(const float* __restrict__ src, _Float16* __restrict__ dst, int n) {
  int i = blockIdx.x * blockDim.x + threadIdx.x;
  if (i < n) dst[i] = (_Float16)src[i];
}

__global__ void route_kernel(const int* __restrict__ labels, int* __restrict__ order,
                             int* __restrict__ navail) {
  int b = threadIdx.x;
  if (b >= kB) return;
  int cnt = 0;
  for (int t = 0; t < kT; ++t)
    if (labels[b * kT + t] != -1) order[b * kT + cnt++] = t;
  navail[b] = cnt;
  for (int t = 0; t < kT; ++t)
    if (labels[b * kT + t] == -1) order[b * kT + cnt++] = t;
}

__global__ void bias_kernel(const int* __restrict__ navail, const unsigned char* __restrict__ tmask,
                            float* __restrict__ bias) {
  int s = blockIdx.x * blockDim.x + threadIdx.x;
  int b = blockIdx.y;
  if (s >= kSP) return;
  float v;
  if (s < kT)            v = (s < navail[b]) ? 0.f : -1e9f;
  else if (s < kT + kNI) v = 0.f;
  else if (s < kS)       v = tmask[b * kNT + (s - kT - kNI)] ? 0.f : -1e9f;
  else                   v = -1e9f;  // padded rows never attended to
  bias[b * kSP + s] = v;
}

__global__ void embed_ln_kernel(const float* __restrict__ img, const float* __restrict__ txt,
                                const float* __restrict__ task, const float* __restrict__ ttask,
                                const float* __restrict__ timg, const float* __restrict__ ttxt,
                                const float* __restrict__ g, const float* __restrict__ bb,
                                const int* __restrict__ order, float* __restrict__ seq) {
  __shared__ float sbuf[8];
  const int s = blockIdx.x, b = blockIdx.y, tid = threadIdx.x;
  const size_t row = ((size_t)b * kSP + s) * kD;
  float x;
  if (s < kT)            x = task[(size_t)order[b * kT + s] * kD + tid] + ttask[tid];
  else if (s < kT + kNI) x = img[((size_t)b * kNI + (s - kT)) * kD + tid] + timg[tid];
  else if (s < kS)       x = txt[((size_t)b * kNT + (s - kT - kNI)) * kD + tid] + ttxt[tid];
  else { seq[row + tid] = 0.f; return; }   // uniform per block
  const float mu = block_reduce_sum_256(x, sbuf) * (1.f / kD);
  const float xc = x - mu;
  const float var = block_reduce_sum_256(xc * xc, sbuf) * (1.f / kD);
  seq[row + tid] = xc * rsqrtf(var + 1e-5f) * g[tid] + bb[tid];
}

__global__ void ln_f16_kernel(const float* __restrict__ in, _Float16* __restrict__ out,
                              const float* __restrict__ g, const float* __restrict__ bb) {
  __shared__ float sbuf[8];
  const int row = blockIdx.x, tid = threadIdx.x;
  const float x = in[(size_t)row * kD + tid];
  const float mu = block_reduce_sum_256(x, sbuf) * (1.f / kD);
  const float xc = x - mu;
  const float var = block_reduce_sum_256(xc * xc, sbuf) * (1.f / kD);
  out[(size_t)row * kD + tid] = (_Float16)(xc * rsqrtf(var + 1e-5f) * g[tid] + bb[tid]);
}

// ---------------- WMMA GEMM:  C[M,N] = A[M,K] @ W[N,K]^T + bias ----------------
// MODE 0: store f16      MODE 1: residual add into f32      MODE 2: exact GELU, store f16
template <int MODE>
__global__ void gemm_wmma_kernel(const _Float16* __restrict__ A, const _Float16* __restrict__ W,
                                 const float* __restrict__ bias, _Float16* __restrict__ out16,
                                 float* __restrict__ out32, int M, int N, int K, int ldo) {
  const int wid  = (blockIdx.x * blockDim.x + threadIdx.x) >> 5;
  const int lane = threadIdx.x & 31;
  const int ngroups = N >> 6;          // 4 n-tiles per wave
  const int mtiles  = M >> 4;
  if (wid >= mtiles * ngroups) return;
  const int mtile = wid / ngroups, ng = wid % ngroups;
  const int half = lane >> 4, l16 = lane & 15;
  const int m0 = mtile * 16;
  const _Float16* arow = A + (size_t)(m0 + l16) * K;

  v8f acc[4] = {};
  const int ktiles = K >> 5;
  for (int kt = 0; kt < ktiles; ++kt) {
    const v16h a = load_a_frag(arow + kt * 32, half);
    #pragma unroll
    for (int j = 0; j < 4; ++j) {
      const int n = (ng * 4 + j) * 16 + l16;
      const v16h bf = *(const v16h*)(W + (size_t)n * K + kt * 32 + half * 16);
      acc[j] = __builtin_amdgcn_wmma_f32_16x16x32_f16(false, a, false, bf, (short)0, acc[j],
                                                      false, false);
    }
  }
  #pragma unroll
  for (int j = 0; j < 4; ++j) {
    const int n = (ng * 4 + j) * 16 + l16;
    const float bn = bias ? bias[n] : 0.f;
    #pragma unroll
    for (int v = 0; v < 8; ++v) {
      const int m = m0 + half * 8 + v;      // C layout: lane=col, VGPR v = row (+8 for upper half)
      const float val = acc[j][v] + bn;
      if (MODE == 0) {
        out16[(size_t)m * ldo + n] = (_Float16)val;
      } else if (MODE == 1) {
        out32[(size_t)m * ldo + n] += val;
      } else {
        const float ge = 0.5f * val * (1.f + erff(val * 0.70710678118654752f));
        out16[(size_t)m * ldo + n] = (_Float16)ge;
      }
    }
  }
}

// ---------------- flash attention: 1 wave per (qtile, head, batch) ----------------
__global__ __launch_bounds__(32) void attn_kernel(const _Float16* __restrict__ qkv,
                                                  const float* __restrict__ bias,
                                                  _Float16* __restrict__ o16) {
  __shared__ __align__(64) _Float16 Pbuf[16 * 40];  // 16 queries x 32 keys, stride 40 (bank-safe)
  __shared__ __align__(64) _Float16 Vbuf[32 * 32];  // 32 keys x 32 dims

  const int lane = threadIdx.x;
  const int half = lane >> 4, l16 = lane & 15;
  const int qt = blockIdx.x, h = blockIdx.y, b = blockIdx.z;
  const size_t rowbase = (size_t)b * kSP;
  const int abase = half ? 16 : 0;

  // Q fragment (A, 16x32 over head dim)
  const _Float16* qrow = qkv + (rowbase + qt * 16 + l16) * 768 + h * kHD;
  v16h qa;
  ((v8h*)&qa)[0] = *(const v8h*)(qrow + abase);
  ((v8h*)&qa)[1] = *(const v8h*)(qrow + abase + 8);

  float rm[8], rl[8];
  v8f c0 = {}, c1 = {};
  #pragma unroll
  for (int v = 0; v < 8; ++v) { rm[v] = -3.0e38f; rl[v] = 0.f; }
  const float scale = 0.17677669529663687f;  // 1/sqrt(32)

  for (int kc = 0; kc < kSP / 32; ++kc) {
    const int k0 = kc * 32;
    // K fragments (B, 32(hd) x 16(keys)): lane = key, half selects hd 0..15 / 16..31
    const _Float16* krow0 = qkv + (rowbase + k0 + l16) * 768 + kD + h * kHD + half * 16;
    const v16h kb0 = *(const v16h*)krow0;
    const v16h kb1 = *(const v16h*)(krow0 + 16 * 768);
    v8f s0 = {}, s1 = {};
    s0 = __builtin_amdgcn_wmma_f32_16x16x32_f16(false, qa, false, kb0, (short)0, s0, false, false);
    s1 = __builtin_amdgcn_wmma_f32_16x16x32_f16(false, qa, false, kb1, (short)0, s1, false, false);

    const float bk0 = bias[rowbase + k0 + l16];
    const float bk1 = bias[rowbase + k0 + 16 + l16];

    float p0a[8], p1a[8];
    #pragma unroll
    for (int v = 0; v < 8; ++v) {
      float x0 = s0[v] * scale + bk0;
      float x1 = s1[v] * scale + bk1;
      float mx = fmaxf(x0, x1);
      #pragma unroll
      for (int msk = 1; msk <= 8; msk <<= 1) mx = fmaxf(mx, __shfl_xor(mx, msk, 32));
      const float nm = fmaxf(rm[v], mx);
      const float alpha = __expf(rm[v] - nm);
      const float e0 = __expf(x0 - nm);
      const float e1 = __expf(x1 - nm);
      float rs = e0 + e1;
      #pragma unroll
      for (int msk = 1; msk <= 8; msk <<= 1) rs += __shfl_xor(rs, msk, 32);
      rl[v] = rl[v] * alpha + rs;
      rm[v] = nm;
      c0[v] *= alpha;
      c1[v] *= alpha;
      p0a[v] = e0;
      p1a[v] = e1;
    }

    __syncthreads();  // protect LDS WAR vs previous iteration
    // stage P (C layout -> row-major) and V (token-major) in LDS
    #pragma unroll
    for (int v = 0; v < 8; ++v) {
      Pbuf[(half * 8 + v) * 40 + l16]      = (_Float16)p0a[v];
      Pbuf[(half * 8 + v) * 40 + 16 + l16] = (_Float16)p1a[v];
    }
    {
      const _Float16* vrow = qkv + (rowbase + k0 + lane) * 768 + 2 * kD + h * kHD;
      *(v16h*)&Vbuf[lane * 32]      = *(const v16h*)vrow;
      *(v16h*)&Vbuf[lane * 32 + 16] = *(const v16h*)(vrow + 16);
    }
    __syncthreads();

    // P as A fragment (16 queries x 32 keys)
    v16h pa;
    {
      const _Float16* pr = Pbuf + l16 * 40;
      ((v8h*)&pa)[0] = *(const v8h*)(pr + abase);
      ((v8h*)&pa)[1] = *(const v8h*)(pr + abase + 8);
    }
    // V as B fragments (32 keys x 16 dims), dims 0..15 and 16..31
    v16h vb0, vb1;
    #pragma unroll
    for (int j = 0; j < 16; ++j) {
      vb0[j] = Vbuf[(half * 16 + j) * 32 + l16];
      vb1[j] = Vbuf[(half * 16 + j) * 32 + 16 + l16];
    }
    c0 = __builtin_amdgcn_wmma_f32_16x16x32_f16(false, pa, false, vb0, (short)0, c0, false, false);
    c1 = __builtin_amdgcn_wmma_f32_16x16x32_f16(false, pa, false, vb1, (short)0, c1, false, false);
  }

  #pragma unroll
  for (int v = 0; v < 8; ++v) {
    const float inv = 1.f / rl[v];
    const size_t orow = (rowbase + qt * 16 + half * 8 + v) * kD + h * kHD;
    o16[orow + l16]      = (_Float16)(c0[v] * inv);
    o16[orow + 16 + l16] = (_Float16)(c1[v] * inv);
  }
}

// ---------------- output LN + inverse permute + per-task towers ----------------
__global__ void head_kernel(const float* __restrict__ seq, const int* __restrict__ order,
                            const int* __restrict__ labels, const float* __restrict__ og,
                            const float* __restrict__ ob, const float* __restrict__ tw1,
                            const float* __restrict__ tb1, const float* __restrict__ tw2,
                            const float* __restrict__ tb2, float* __restrict__ out) {
  __shared__ float sbuf[8];
  __shared__ float yv[kD];
  __shared__ float hid[kTH];
  const int trow = blockIdx.x, b = blockIdx.y, tid = threadIdx.x;

  const float x = seq[((size_t)b * kSP + trow) * kD + tid];
  const float mu = block_reduce_sum_256(x, sbuf) * (1.f / kD);
  const float xc = x - mu;
  const float var = block_reduce_sum_256(xc * xc, sbuf) * (1.f / kD);
  yv[tid] = xc * rsqrtf(var + 1e-5f) * og[tid] + ob[tid];
  __syncthreads();

  const int tcol = order[b * kT + trow];  // task id this token belongs to
  const int j = tid >> 3, seg = tid & 7;
  const float* wrow = tw1 + ((size_t)tcol * kTH + j) * kD + seg * 32;
  const float* yp = yv + seg * 32;
  float p = 0.f;
  #pragma unroll
  for (int c = 0; c < 32; ++c) p += yp[c] * wrow[c];
  p += __shfl_xor(p, 1, 32);
  p += __shfl_xor(p, 2, 32);
  p += __shfl_xor(p, 4, 32);
  if (seg == 0) hid[j] = fmaxf(p + tb1[tcol * kTH + j], 0.f);
  __syncthreads();
  if (tid == 0) {
    float s = 0.f;
    for (int jj = 0; jj < kTH; ++jj) s += hid[jj] * tw2[tcol * kTH + jj];
    out[(size_t)b * kT + tcol] = (labels[b * kT + tcol] != -1) ? (s + tb2[tcol]) : 0.f;
  }
}

// ---------------- host launch ----------------
extern "C" void kernel_launch(void* const* d_in, const int* in_sizes, int n_in,
                              void* d_out, int out_size, void* d_ws, size_t ws_size,
                              hipStream_t stream) {
  (void)in_sizes; (void)n_in; (void)out_size; (void)ws_size;

  const float* img   = (const float*)d_in[0];
  const float* txt   = (const float*)d_in[1];
  const float* task  = (const float*)d_in[2];
  const float* ttask = (const float*)d_in[3];
  const float* timg  = (const float*)d_in[4];
  const float* ttxt  = (const float*)d_in[5];
  const float* in_g  = (const float*)d_in[6];
  const float* in_b  = (const float*)d_in[7];
  const float* Wqkv  = (const float*)d_in[8];
  const float* bqkv  = (const float*)d_in[9];
  const float* Wo    = (const float*)d_in[10];
  const float* bo    = (const float*)d_in[11];
  const float* ln1g  = (const float*)d_in[12];
  const float* ln1b  = (const float*)d_in[13];
  const float* ln2g  = (const float*)d_in[14];
  const float* ln2b  = (const float*)d_in[15];
  const float* W1    = (const float*)d_in[16];
  const float* b1    = (const float*)d_in[17];
  const float* W2    = (const float*)d_in[18];
  const float* b2    = (const float*)d_in[19];
  const float* outg  = (const float*)d_in[20];
  const float* outb  = (const float*)d_in[21];
  const float* tw1   = (const float*)d_in[22];
  const float* tb1   = (const float*)d_in[23];
  const float* tw2   = (const float*)d_in[24];
  const float* tb2   = (const float*)d_in[25];
  const unsigned char* tmask = (const unsigned char*)d_in[26];
  const int* labels  = (const int*)d_in[27];

  // workspace carve-up (256B aligned)
  char* ws = (char*)d_ws;
  size_t off = 0;
  auto alloc = [&](size_t bytes) -> void* {
    void* p = ws + off;
    off = (off + bytes + 255) & ~(size_t)255;
    return p;
  };
  int*      order  = (int*)alloc((size_t)kB * kT * 4);
  int*      navail = (int*)alloc((size_t)kB * 4);
  float*    biasb  = (float*)alloc((size_t)kB * kSP * 4);
  float*    seq    = (float*)alloc((size_t)kM * kD * 4);
  _Float16* h16    = (_Float16*)alloc((size_t)kM * kD * 2);
  _Float16* qkv16  = (_Float16*)alloc((size_t)kM * 3 * kD * 2);
  _Float16* o16    = (_Float16*)alloc((size_t)kM * kD * 2);
  _Float16* ff16   = (_Float16*)alloc((size_t)kM * kFF * 2);
  _Float16* wq16   = (_Float16*)alloc((size_t)kL * 3 * kD * kD * 2);
  _Float16* wo16   = (_Float16*)alloc((size_t)kL * kD * kD * 2);
  _Float16* w116   = (_Float16*)alloc((size_t)kL * kFF * kD * 2);
  _Float16* w216   = (_Float16*)alloc((size_t)kL * kD * kFF * 2);

  auto cvt = [&](const float* src, _Float16* dst, int n) {
    cvt_f32_f16_kernel<<<(n + 255) / 256, 256, 0, stream>>>(src, dst, n);
  };
  cvt(Wqkv, wq16, kL * 3 * kD * kD);
  cvt(Wo,   wo16, kL * kD * kD);
  cvt(W1,   w116, kL * kFF * kD);
  cvt(W2,   w216, kL * kD * kFF);

  route_kernel<<<1, kB, 0, stream>>>(labels, order, navail);
  bias_kernel<<<dim3((kSP + 255) / 256, kB), 256, 0, stream>>>(navail, tmask, biasb);
  embed_ln_kernel<<<dim3(kSP, kB), 256, 0, stream>>>(img, txt, task, ttask, timg, ttxt,
                                                     in_g, in_b, order, seq);

  auto gemm_grid = [](int M, int N) { return ((M / 16) * (N / 64) + 7) / 8; };

  for (int l = 0; l < kL; ++l) {
    ln_f16_kernel<<<kM, 256, 0, stream>>>(seq, h16, ln1g + l * kD, ln1b + l * kD);
    gemm_wmma_kernel<0><<<gemm_grid(kM, 3 * kD), 256, 0, stream>>>(
        h16, wq16 + (size_t)l * 3 * kD * kD, bqkv + l * 3 * kD, qkv16, nullptr,
        kM, 3 * kD, kD, 3 * kD);
    attn_kernel<<<dim3(kSP / 16, kNH, kB), 32, 0, stream>>>(qkv16, biasb, o16);
    gemm_wmma_kernel<1><<<gemm_grid(kM, kD), 256, 0, stream>>>(
        o16, wo16 + (size_t)l * kD * kD, bo + l * kD, nullptr, seq, kM, kD, kD, kD);
    ln_f16_kernel<<<kM, 256, 0, stream>>>(seq, h16, ln2g + l * kD, ln2b + l * kD);
    gemm_wmma_kernel<2><<<gemm_grid(kM, kFF), 256, 0, stream>>>(
        h16, w116 + (size_t)l * kFF * kD, b1 + l * kFF, ff16, nullptr, kM, kFF, kD, kFF);
    gemm_wmma_kernel<1><<<gemm_grid(kM, kD), 256, 0, stream>>>(
        ff16, w216 + (size_t)l * kD * kFF, b2 + l * kD, nullptr, seq, kM, kD, kFF, kD);
  }

  head_kernel<<<dim3(kT, kB), 256, 0, stream>>>(seq, order, labels, outg, outb,
                                                tw1, tb1, tw2, tb2, (float*)d_out);
}
